// Attention1D_68212670595445
// MI455X (gfx1250) — matrix operand order repaired and checked by reading, not compiled, not run on hardware
//
#include <hip/hip_runtime.h>
#include <hip/hip_bf16.h>

#define BS   256
#define NSEQ 512
#define QD   2048
#define VD   2048
#define AD   1024
#define MTOT (BS * NSEQ)   // 131072

// GEMM tiling: block 128x128, 8 waves as 4(M) x 2(N), wave tile 32x64 (2x4 WMMA)
#define BM 128
#define BN 128
#define BK 32
#define BKF (BK + 4)       // A tile kept as f32: 36 floats = 144B stride (16B aligned, conflict-free)
#define LDB (BK + 8)       // B tile bf16: 40 ushorts = 80B stride  (16B aligned, conflict-free)
#define KD  2048           // K dim (QD == VD)
#define NSTAGE (KD / BK)   // 64

typedef __attribute__((ext_vector_type(16))) __bf16        v16bf;
typedef __attribute__((ext_vector_type(4)))  __bf16        v4bf;
typedef __attribute__((ext_vector_type(8)))  float         v8f;
typedef __attribute__((ext_vector_type(4)))  float         v4f;
typedef __attribute__((ext_vector_type(4)))  unsigned int  u32x4;
typedef int v4i __attribute__((vector_size(4 * sizeof(int))));  // matches builtin param type

union Frag16 {
    v16bf bf;
    u32x4 q[2];
};
union FragA {
    v16bf bf;
    v4bf  h[4];
};

// -------- async global->LDS copy (CDNA5 ASYNCcnt path), sync fallback --------
#if defined(__has_builtin)
#  if __has_builtin(__builtin_amdgcn_global_load_async_to_lds_b128) && \
      __has_builtin(__builtin_amdgcn_s_wait_asynccnt)
#    define USE_ASYNC_LDS 1
#  endif
#endif
#ifndef USE_ASYNC_LDS
#  define USE_ASYNC_LDS 0
#endif

__device__ __forceinline__ void copy16_g2l(const void* g, void* l) {
#if USE_ASYNC_LDS
    __builtin_amdgcn_global_load_async_to_lds_b128(
        (v4i*)(void*)g, (v4i*)l, 0, 0);
#else
    *reinterpret_cast<u32x4*>(l) = *reinterpret_cast<const u32x4*>(g);
#endif
}
__device__ __forceinline__ void wait_async_prev() {   // keep next tile (6 ops) in flight
#if USE_ASYNC_LDS
    __builtin_amdgcn_s_wait_asynccnt(6);
#endif
}
__device__ __forceinline__ void wait_async_all() {
#if USE_ASYNC_LDS
    __builtin_amdgcn_s_wait_asynccnt(0);
#endif
}

__device__ __forceinline__ unsigned short f32_to_bf16(float f) {
    __bf16 h = (__bf16)f;
    return __builtin_bit_cast(unsigned short, h);
}

// ---------------------------------------------------------------------------
// Shared WMMA main loop: acc[2][4] (+=) A[rowBase..+128, :2048] * BT[colBase..+128, :2048]^T
// A: f32 row-major (K stride 2048), BT: bf16 row-major [col][K] (pre-transposed)
// ---------------------------------------------------------------------------
__device__ __forceinline__ void gemm_mainloop(
    const float* __restrict__ Aglob,
    const unsigned short* __restrict__ BTglob,
    int rowBase, int colBase,
    float (&Af)[2][BM][BKF],
    unsigned short (&Bsh)[2][BN][LDB],
    v8f (&acc)[2][4])
{
    const int tid  = threadIdx.x;
    const int lane = tid & 31;
    const int wave = tid >> 5;
    const int wm   = wave >> 1;          // 0..3  (32-row strip)
    const int wn   = wave & 1;           // 0..1  (64-col strip)
    const int l15  = lane & 15;
    const int hi   = lane >> 4;
    const int kA0  = hi ? 8 : 0;         // ISA 16-bit A 16x32 lane->K mapping
    const int kB0  = hi ? 8 : 0;         // B 32x16 lane->K mapping (same as A)

    auto issue = [&](int k0, int buf) {
#pragma unroll
        for (int i = 0; i < 4; ++i) {    // A: 128x32 f32 (16B per lane-op)
            int idx = tid + i * 256;
            int r = idx >> 3, c4 = (idx & 7) * 4;
            copy16_g2l(&Aglob[(size_t)(rowBase + r) * KD + k0 + c4], &Af[buf][r][c4]);
        }
#pragma unroll
        for (int i = 0; i < 2; ++i) {    // B: 128x32 bf16
            int idx = tid + i * 256;
            int nr = idx >> 2, c8 = (idx & 3) * 8;
            copy16_g2l(&BTglob[(size_t)(colBase + nr) * KD + k0 + c8], &Bsh[buf][nr][c8]);
        }
    };

    issue(0, 0);
    for (int s = 0; s < NSTAGE; ++s) {
        const int buf = s & 1;
        if (s + 1 < NSTAGE) { issue((s + 1) * BK, buf ^ 1); wait_async_prev(); }
        else                { wait_async_all(); }
        __syncthreads();                 // tile s visible to all waves

        FragA  a[2];
        Frag16 b[4];
#pragma unroll
        for (int tm = 0; tm < 2; ++tm) { // A frags: load f32, convert in-register
            const float* p = &Af[buf][wm * 32 + tm * 16 + l15][0];
            v4f f0 = *reinterpret_cast<const v4f*>(p + kA0);
            v4f f1 = *reinterpret_cast<const v4f*>(p + kA0 + 4);
            v4f f2 = *reinterpret_cast<const v4f*>(p + kA0 + 16);
            v4f f3 = *reinterpret_cast<const v4f*>(p + kA0 + 20);
            a[tm].h[0] = __builtin_convertvector(f0, v4bf);
            a[tm].h[1] = __builtin_convertvector(f1, v4bf);
            a[tm].h[2] = __builtin_convertvector(f2, v4bf);
            a[tm].h[3] = __builtin_convertvector(f3, v4bf);
        }
#pragma unroll
        for (int tn = 0; tn < 4; ++tn) {
            const unsigned short* p = &Bsh[buf][wn * 64 + tn * 16 + l15][0];
            b[tn].q[0] = *reinterpret_cast<const u32x4*>(p + kB0);
            b[tn].q[1] = *reinterpret_cast<const u32x4*>(p + kB0 + 16);
        }
#pragma unroll
        for (int tm = 0; tm < 2; ++tm)
#pragma unroll
            for (int tn = 0; tn < 4; ++tn)
                acc[tm][tn] = __builtin_amdgcn_wmma_f32_16x16x32_bf16(
                    false, a[tm].bf, false, b[tn].bf,
                    (short)0, acc[tm][tn], false, false);
        __syncthreads();                 // all waves done with buf before it is refilled
    }
}

// ---------------------------------------------------------------------------
// Kernel 1: transpose + convert weights to bf16: [K][A] f32 -> [A][K] bf16
// ---------------------------------------------------------------------------
__global__ __launch_bounds__(256)
void transpose_bf16_kernel(const float* __restrict__ Wvis,
                           const float* __restrict__ Wques,
                           unsigned short* __restrict__ WvisT,
                           unsigned short* __restrict__ WquesT) {
    __shared__ unsigned short t[64][65];
    const float* src = blockIdx.z ? Wques : Wvis;
    unsigned short* dst = blockIdx.z ? WquesT : WvisT;
    const int a0 = blockIdx.x * 64;
    const int v0 = blockIdx.y * 64;
#pragma unroll
    for (int i = 0; i < 16; ++i) {
        int idx = threadIdx.x + i * 256;
        int r = idx >> 6, c = idx & 63;
        t[c][r] = f32_to_bf16(src[(size_t)(v0 + r) * AD + (a0 + c)]);
    }
    __syncthreads();
#pragma unroll
    for (int i = 0; i < 16; ++i) {
        int idx = threadIdx.x + i * 256;
        int r = idx >> 6, c = idx & 63;
        dst[(size_t)(a0 + r) * VD + (v0 + c)] = t[r][c];
    }
}

// ---------------------------------------------------------------------------
// Kernel 2: X2att = X @ W_ques + b_ques   (WMMA bf16, M=256 N=1024 K=2048)
// ---------------------------------------------------------------------------
__global__ __launch_bounds__(256)
void x2att_gemm_kernel(const float* __restrict__ X,
                       const unsigned short* __restrict__ WquesT,
                       const float* __restrict__ bques,
                       float* __restrict__ X2att) {
    __shared__ float          Af[2][BM][BKF];
    __shared__ unsigned short Bsh[2][BN][LDB];
    __shared__ float bq[BN];

    const int tid  = threadIdx.x;
    const int lane = tid & 31;
    const int wave = tid >> 5;
    const int wm   = wave >> 1;
    const int wn   = wave & 1;
    const int l15  = lane & 15;
    const int hi   = lane >> 4;

    const int rowBase = blockIdx.y * BM;
    const int colBase = blockIdx.x * BN;

    if (tid < BN) bq[tid] = bques[colBase + tid];

    v8f acc[2][4] = {};
    gemm_mainloop(X, WquesT, rowBase, colBase, Af, Bsh, acc);

#pragma unroll
    for (int tm = 0; tm < 2; ++tm)
#pragma unroll
        for (int tn = 0; tn < 4; ++tn) {
            int colL = wn * 64 + tn * 16 + l15;
            int col  = colBase + colL;
            int row0 = rowBase + wm * 32 + tm * 16 + hi * 8;
            float bb = bq[colL];
#pragma unroll
            for (int r = 0; r < 8; ++r)
                X2att[(size_t)(row0 + r) * AD + col] = acc[tm][tn][r] + bb;
        }
}

// ---------------------------------------------------------------------------
// Kernel 3: zero logits
// ---------------------------------------------------------------------------
__global__ void zero_kernel(float* __restrict__ p, int n) {
    int i = blockIdx.x * 256 + threadIdx.x;
    if (i < n) p[i] = 0.0f;
}

// ---------------------------------------------------------------------------
// Kernel 4 (main): fused  Y @ W_vis -> +X2att -> relu -> dot(W_map) -> logits
// grid = (AD/BN = 8 column chunks, MTOT/BM = 1024 row blocks)
// ---------------------------------------------------------------------------
__global__ __launch_bounds__(256)
void fused_logits_kernel(const float* __restrict__ Y,
                         const unsigned short* __restrict__ WvisT,
                         const float* __restrict__ X2att,
                         const float* __restrict__ Wmap,
                         float* __restrict__ logits) {
    __shared__ float          Af[2][BM][BKF];
    __shared__ unsigned short Bsh[2][BN][LDB];
    __shared__ float xrow[BN];
    __shared__ float wms[BN];

    const int tid  = threadIdx.x;
    const int lane = tid & 31;
    const int wave = tid >> 5;
    const int wm   = wave >> 1;
    const int wn   = wave & 1;
    const int l15  = lane & 15;
    const int hi   = lane >> 4;

    const int rowBase = blockIdx.y * BM;     // global M row (= b*512 + n)
    const int colBase = blockIdx.x * BN;
    const int bidx    = rowBase >> 9;        // 512 rows/batch, BM=128 divides

    if (tid < BN) {
        xrow[tid] = X2att[(size_t)bidx * AD + colBase + tid];
        wms[tid]  = Wmap[colBase + tid];
    }

    v8f acc[2][4] = {};
    gemm_mainloop(Y, WvisT, rowBase, colBase, Af, Bsh, acc);

    // relu(acc + X2att[b,col]) * W_map[col]; reduce 4 in-register cols +
    // 16 cross-lane cols; atomic-add partial logits.
#pragma unroll
    for (int tm = 0; tm < 2; ++tm) {
        float s[8];
#pragma unroll
        for (int r = 0; r < 8; ++r) s[r] = 0.0f;
#pragma unroll
        for (int tn = 0; tn < 4; ++tn) {
            int colL = wn * 64 + tn * 16 + l15;
            float xb = xrow[colL];
            float wv = wms[colL];
#pragma unroll
            for (int r = 0; r < 8; ++r) {
                float v = fmaxf(acc[tm][tn][r] + xb, 0.0f);
                s[r] = fmaf(v, wv, s[r]);
            }
        }
#pragma unroll
        for (int r = 0; r < 8; ++r) {
            float t = s[r];
            t += __shfl_xor(t, 1);
            t += __shfl_xor(t, 2);
            t += __shfl_xor(t, 4);
            t += __shfl_xor(t, 8);
            s[r] = t;
        }
        if (l15 == 0) {                   // lanes 0 / 16 own rows +0 / +8
            int row0 = rowBase + wm * 32 + tm * 16 + hi * 8;
#pragma unroll
            for (int r = 0; r < 8; ++r)
                atomicAdd(&logits[row0 + r], s[r]);
        }
    }
}

// ---------------------------------------------------------------------------
// Kernel 5: softmax over n (per batch row)
// ---------------------------------------------------------------------------
__global__ __launch_bounds__(512)
void softmax_kernel(const float* __restrict__ logits, float* __restrict__ w) {
    __shared__ float sm[512];
    const int b = blockIdx.x, tid = threadIdx.x;
    float v = logits[b * NSEQ + tid];
    sm[tid] = v;
    __syncthreads();
    for (int s = 256; s > 0; s >>= 1) {
        if (tid < s) sm[tid] = fmaxf(sm[tid], sm[tid + s]);
        __syncthreads();
    }
    const float mx = sm[0];
    __syncthreads();
    const float e = __expf(v - mx);
    sm[tid] = e;
    __syncthreads();
    for (int s = 256; s > 0; s >>= 1) {
        if (tid < s) sm[tid] += sm[tid + s];
        __syncthreads();
    }
    w[b * NSEQ + tid] = e / sm[0];
}

// ---------------------------------------------------------------------------
// Kernel 6: Y_attended[b,v] = sum_n w[b,n] * Y[b,n,v]  (bandwidth bound)
// ---------------------------------------------------------------------------
__global__ __launch_bounds__(256)
void attend_kernel(const float* __restrict__ Y, const float* __restrict__ w,
                   float* __restrict__ out) {
    __shared__ float wsm[NSEQ];
    const int b   = blockIdx.y;
    const int col = blockIdx.x * 1024 + threadIdx.x * 4;
    for (int i = threadIdx.x; i < NSEQ; i += 256) wsm[i] = w[b * NSEQ + i];
    __syncthreads();
    const float* Yb = Y + (size_t)b * NSEQ * VD + col;
    float a0 = 0.f, a1 = 0.f, a2 = 0.f, a3 = 0.f;
#pragma unroll 4
    for (int n = 0; n < NSEQ; ++n) {
        const float4 y = *reinterpret_cast<const float4*>(Yb + (size_t)n * VD);
        const float wv = wsm[n];
        a0 = fmaf(wv, y.x, a0);
        a1 = fmaf(wv, y.y, a1);
        a2 = fmaf(wv, y.z, a2);
        a3 = fmaf(wv, y.w, a3);
    }
    float4 o; o.x = a0; o.y = a1; o.z = a2; o.w = a3;
    *reinterpret_cast<float4*>(&out[(size_t)b * VD + col]) = o;
}

// ---------------------------------------------------------------------------
extern "C" void kernel_launch(void* const* d_in, const int* in_sizes, int n_in,
                              void* d_out, int out_size, void* d_ws, size_t ws_size,
                              hipStream_t stream) {
    const float* X     = (const float*)d_in[0];   // [256, 2048]
    const float* Y     = (const float*)d_in[1];   // [256, 512, 2048]
    const float* Wvis  = (const float*)d_in[2];   // [2048, 1024]
    const float* Wques = (const float*)d_in[3];   // [2048, 1024]
    const float* bques = (const float*)d_in[4];   // [1024]
    const float* Wmap  = (const float*)d_in[5];   // [1024]
    // d_in[6] = b_map: constant shift before softmax -> cancels, ignored.
    float* out = (float*)d_out;                   // [256, 2048]

    char* ws = (char*)d_ws;
    unsigned short* WvisT  = (unsigned short*)(ws);                           // 4 MB
    unsigned short* WquesT = (unsigned short*)(ws + (size_t)4 * 1024 * 1024); // 4 MB
    float* X2att  = (float*)(ws + (size_t)8 * 1024 * 1024);                   // 1 MB
    float* logits = (float*)(ws + (size_t)9 * 1024 * 1024);                   // 512 KB
    float* watt   = (float*)(ws + (size_t)9 * 1024 * 1024 + 512 * 1024);      // 512 KB

    transpose_bf16_kernel<<<dim3(AD / 64, VD / 64, 2), 256, 0, stream>>>(
        Wvis, Wques, WvisT, WquesT);
    x2att_gemm_kernel<<<dim3(AD / BN, BS / BM), 256, 0, stream>>>(
        X, WquesT, bques, X2att);
    zero_kernel<<<(MTOT + 255) / 256, 256, 0, stream>>>(logits, MTOT);
    fused_logits_kernel<<<dim3(AD / BN, MTOT / BM), 256, 0, stream>>>(
        Y, WvisT, X2att, Wmap, logits);
    softmax_kernel<<<BS, NSEQ, 0, stream>>>(logits, watt);
    attend_kernel<<<dim3(2, BS), 256, 0, stream>>>(Y, watt, out);
}
